// SparseModel_13812614824201
// MI455X (gfx1250) — matrix operand order repaired and checked by reading, not compiled, not run on hardware
//
#include <hip/hip_runtime.h>
#include <stdint.h>

// ---- CDNA5 vector types ----
typedef __attribute__((ext_vector_type(16))) __bf16 v16bf;
typedef __attribute__((ext_vector_type(8)))  float  v8f;
typedef __attribute__((ext_vector_type(4)))  unsigned int u32x4;
typedef __attribute__((ext_vector_type(8)))  int i32x8;
typedef __attribute__((ext_vector_type(4)))  int i32x4;

// gcc-style vector + addrspace pointers for the async-to-LDS builtin
typedef int v4i_vs __attribute__((vector_size(16)));
typedef __attribute__((address_space(1))) v4i_vs* as1_v4i_ptr;
typedef __attribute__((address_space(3))) v4i_vs* as3_v4i_ptr;

#define CIN    256
#define COUT   256
#define HH     56
#define WW     56
#define HW     3136
#define NPIX   100352            // 32*56*56
#define KTOT   2304              // 9 * 256 ; reordered k' = rs*256 + ci
#define BK     32
#define NCHUNK 72                // 2304/32
#define SA     36                // A LDS row stride (shorts): 64B data + 8B TDM pad
#define SB     40                // B LDS col stride (shorts): keeps b128 stores 16B-aligned
#define XSP    (NPIX * CIN)      // shorts per x split plane
#define WSP    (COUT * KTOT)     // shorts per w split plane
#define ASLOT  (128 * SA)        // shorts per A plane tile in LDS
#define BSLOT  (128 * SB)        // shorts per B plane tile in LDS

#if defined(__has_builtin)
#if __has_builtin(__builtin_amdgcn_tensor_load_to_lds)
#define HAVE_TDM 1
#endif
#if __has_builtin(__builtin_amdgcn_global_load_async_to_lds_b128)
#define HAVE_ASYNC 1
#endif
#endif
#ifndef HAVE_TDM
#define HAVE_TDM 0
#endif
#ifndef HAVE_ASYNC
#define HAVE_ASYNC 0
#endif

__device__ __forceinline__ void wait_asynccnt0() {
#if defined(__has_builtin)
#if __has_builtin(__builtin_amdgcn_s_wait_asynccnt)
  __builtin_amdgcn_s_wait_asynccnt(0);
#else
  asm volatile("s_wait_asynccnt 0x0" ::: "memory");
#endif
#else
  asm volatile("s_wait_asynccnt 0x0" ::: "memory");
#endif
}

union FragB16 { v16bf v; uint32_t u[8]; };

__device__ __forceinline__ unsigned short f32_to_bf16_rne(float f) {
  uint32_t u = __builtin_bit_cast(uint32_t, f);
  uint32_t r = u + 0x7FFFu + ((u >> 16) & 1u);
  return (unsigned short)(r >> 16);
}
__device__ __forceinline__ float bf16_to_f32(unsigned short h) {
  return __builtin_bit_cast(float, (uint32_t)h << 16);
}

// A-fragment (16x32 bf16, MxK), LDS [row][k] stride SA, k contiguous.
__device__ __forceinline__ v16bf load_frag_a(const unsigned short* lds, int rowBase, int lane) {
  FragB16 f;
  const int m = lane & 15;
  const int half = lane >> 4;
  const unsigned short* rp = lds + (rowBase + m) * SA;
#pragma unroll
  for (int v = 0; v < 4; ++v)
    f.u[v] = *(const uint32_t*)(rp + half * 8 + 2 * v);
#pragma unroll
  for (int v = 0; v < 4; ++v)
    f.u[4 + v] = *(const uint32_t*)(rp + 16 + half * 8 + 2 * v);
  return f.v;
}

// B-fragment (32x16 bf16, KxN), LDS [col][k] stride SB, k contiguous.
__device__ __forceinline__ v16bf load_frag_b(const unsigned short* lds, int colBase, int lane) {
  FragB16 f;
  const int n = lane & 15;
  const int half = lane >> 4;
  const unsigned short* cp = lds + (colBase + n) * SB;
#pragma unroll
  for (int v = 0; v < 4; ++v)
    f.u[v] = *(const uint32_t*)(cp + half * 8 + 2 * v);
#pragma unroll
  for (int v = 0; v < 4; ++v)
    f.u[4 + v] = *(const uint32_t*)(cp + 16 + half * 8 + 2 * v);
  return f.v;
}

// ================= preprocessing =================
// x: NCHW f32 -> xs: [2(hi/lo)][NPIX][CIN] bf16 (NHWC)
__global__ __launch_bounds__(256) void prep_x(const float* __restrict__ x,
                                              unsigned short* __restrict__ xs) {
  const int pix = blockIdx.x;
  const int ci  = threadIdx.x;
  const int n   = pix / HW;
  const int hw  = pix - n * HW;
  float f = x[(size_t)(n * CIN + ci) * HW + hw];
  unsigned short hi = f32_to_bf16_rne(f);
  unsigned short lo = f32_to_bf16_rne(f - bf16_to_f32(hi));
  xs[(size_t)pix * CIN + ci] = hi;
  xs[(size_t)XSP + (size_t)pix * CIN + ci] = lo;
}

// w: [COUT][CIN][3][3] f32 -> wsp: [2(hi/lo)][COUT][k'=rs*256+ci] bf16
// Also zeroes the 64B zero-pad block used by async OOB redirection.
__global__ __launch_bounds__(256) void prep_w(const float* __restrict__ w,
                                              unsigned short* __restrict__ wsp,
                                              unsigned short* __restrict__ zpad) {
  const int co = blockIdx.x;
  const int ci = threadIdx.x;
  if (co == 0 && ci < 32) zpad[ci] = 0;
#pragma unroll
  for (int rs = 0; rs < 9; ++rs) {
    float f = w[(size_t)co * KTOT + ci * 9 + rs];
    unsigned short hi = f32_to_bf16_rne(f);
    unsigned short lo = f32_to_bf16_rne(f - bf16_to_f32(hi));
    wsp[(size_t)co * KTOT + rs * CIN + ci] = hi;
    wsp[(size_t)WSP + (size_t)co * KTOT + rs * CIN + ci] = lo;
  }
}

// ================= main implicit-GEMM =================
__global__ __launch_bounds__(256)
void conv_gemm(const unsigned short* __restrict__ xs,
               const unsigned short* __restrict__ wsp,
               const unsigned short* __restrict__ zpad,
               float* __restrict__ out) {
  __shared__ unsigned short lsA[2][2 * ASLOT];  // [buf][plane][row][k]
  __shared__ unsigned short lsB[2][2 * BSLOT];  // [buf][plane][col][k]

  const int t    = threadIdx.x;
  const int lane = t & 31;
  const int wave = t >> 5;
  const int waveRow = wave & 3;    // 4 row groups x 32 rows
  const int waveCol = wave >> 2;   // 2 col groups x 64 cols
  const int blockCol = blockIdx.x; // 784 pixel tiles
  const int blockRow = blockIdx.y; // 2 cout tiles

  // B staging: per thread 4 x 16B tasks: planes {hi,lo} x cols {c0, c0+64}, kgroup t&3
  const int bkg = t & 3;
  int bn[2], boh[2], bow[2];
#pragma unroll
  for (int cs = 0; cs < 2; ++cs) {
    const int col = (t >> 2) + 64 * cs;
    const int p   = blockCol * 128 + col;
    bn[cs]  = p / HW;
    const int hw = p - bn[cs] * HW;
    boh[cs] = hw / WW;
    bow[cs] = hw - boh[cs] * WW;
  }

#if HAVE_ASYNC
  // Async gather: each lane DMA-copies 16B straight into LDS (ASYNCcnt).
  // OOB taps redirect the source address to a zeroed 64B block.
  auto stage_b_async = [&](int c, int bufn) {
    const int kc  = c * BK;
    const int rs  = kc >> 8;
    const int r   = rs / 3;
    const int s   = rs - r * 3;
    const int ci0 = kc & 255;
#pragma unroll
    for (int j = 0; j < 4; ++j) {
      const int plane = j >> 1, cs = j & 1;
      const int col = (t >> 2) + 64 * cs;
      const int ih = boh[cs] + r - 1;  // PAD=1
      const int iw = bow[cs] + s - 1;
      const bool ok = (unsigned)ih < (unsigned)HH && (unsigned)iw < (unsigned)WW;
      const size_t src = (size_t)plane * XSP +
          (size_t)((bn[cs] * HH + ih) * WW + iw) * CIN + (ci0 + bkg * 8);
      const unsigned short* gp = ok ? (xs + src) : zpad;
      unsigned short* lp = &lsB[bufn][plane * BSLOT + col * SB + bkg * 8];
      __builtin_amdgcn_global_load_async_to_lds_b128(
          (as1_v4i_ptr)const_cast<unsigned short*>(gp),
          (as3_v4i_ptr)lp, 0, 0);
    }
  };
#else
  auto stage_b_load = [&](int c, uint4* breg) {
    const int kc  = c * BK;
    const int rs  = kc >> 8;
    const int r   = rs / 3;
    const int s   = rs - r * 3;
    const int ci0 = kc & 255;
#pragma unroll
    for (int j = 0; j < 4; ++j) {
      const int plane = j >> 1, cs = j & 1;
      const int ih = boh[cs] + r - 1;  // PAD=1
      const int iw = bow[cs] + s - 1;
      if ((unsigned)ih < (unsigned)HH && (unsigned)iw < (unsigned)WW) {
        const size_t src = (size_t)plane * XSP +
            (size_t)((bn[cs] * HH + ih) * WW + iw) * CIN + (ci0 + bkg * 8);
        breg[j] = *(const uint4*)(xs + src);
      } else {
        breg[j] = make_uint4(0u, 0u, 0u, 0u);
      }
    }
  };
  auto stage_b_store = [&](int bufn, const uint4* breg) {
#pragma unroll
    for (int j = 0; j < 4; ++j) {
      const int plane = j >> 1, cs = j & 1;
      const int col = (t >> 2) + 64 * cs;
      *(uint4*)&lsB[bufn][plane * BSLOT + col * SB + bkg * 8] = breg[j];
    }
  };
#endif

#if HAVE_TDM
  // One 3D TDM per chunk: dim0 = 32 k (contig), dim1 = 128 co (stride 2304),
  // dim2 = 2 planes (stride 256*2304). LDS pad: 8B after every 64B -> row stride 72B.
  auto tdm_issue_a = [&](int c, int bufn) {
    const int kc = c * BK;
    const unsigned long long ga = (unsigned long long)(uintptr_t)wsp +
        (unsigned long long)((size_t)blockRow * 128 * KTOT + kc) * 2ull;
    const unsigned ldsOff = (unsigned)(uintptr_t)(&lsA[bufn][0]);
    u32x4 g0;
    g0[0] = 1u;                                   // count=1, load, user mode
    g0[1] = ldsOff;                               // lds_addr
    g0[2] = (unsigned)ga;                         // global_addr lo
    g0[3] = ((unsigned)(ga >> 32) & 0x1FFFFFFu) | (2u << 30);  // addr hi | type=2
    i32x8 g1;
    g1[0] = (1 << 16) | (1 << 20) | (3 << 22) | (1 << 25); // 2B elems, pad on, 64B/8B
    g1[1] = (int)(2304u << 16);                   // tensor_dim0 = 2304 (lo16 in [31:16])
    g1[2] = (int)(256u << 16);                    // dim0 hi16=0 | tensor_dim1=256 lo16
    g1[3] = (int)(32u << 16);                     // dim1 hi16=0 | tile_dim0 = 32
    g1[4] = (int)(128u | (2u << 16));             // tile_dim1 = 128, tile_dim2 = 2
    g1[5] = 2304;                                 // tensor_dim0_stride lo32
    g1[6] = 0;                                    // stride0 hi16 | stride1 lo16 (589824 lo16=0)
    g1[7] = 9;                                    // tensor_dim1_stride hi (589824>>16)
    i32x4 g2; g2[0] = 2; g2[1] = 0; g2[2] = 0; g2[3] = 0;   // tensor_dim2 = 2
    i32x4 g3; g3[0] = 0; g3[1] = 0; g3[2] = 0; g3[3] = 0;
#if defined(__clang_major__) && __clang_major__ >= 23
    i32x8 g4;
#pragma unroll
    for (int i = 0; i < 8; ++i) g4[i] = 0;
    __builtin_amdgcn_tensor_load_to_lds(g0, g1, g2, g3, g4, 0);
#else
    __builtin_amdgcn_tensor_load_to_lds(g0, g1, g2, g3, 0);
#endif
  };
#else
  auto stage_a_load = [&](int c, uint4* areg) {
    const int kc = c * BK;
#pragma unroll
    for (int j = 0; j < 4; ++j) {
      const int plane = j >> 1, cs = j & 1;
      const int row = (t >> 2) + 64 * cs;
      const size_t src = (size_t)plane * WSP +
          (size_t)(blockRow * 128 + row) * KTOT + kc + bkg * 8;
      areg[j] = *(const uint4*)(wsp + src);
    }
  };
  auto stage_a_store = [&](int bufn, const uint4* areg) {
#pragma unroll
    for (int j = 0; j < 4; ++j) {
      const int plane = j >> 1, cs = j & 1;
      const int row = (t >> 2) + 64 * cs;
      uint2* dst = (uint2*)&lsA[bufn][plane * ASLOT + row * SA + bkg * 8];
      dst[0] = make_uint2(areg[j].x, areg[j].y);
      dst[1] = make_uint2(areg[j].z, areg[j].w);
    }
  };
#endif

  v8f acc[2][4];
#pragma unroll
  for (int i = 0; i < 2; ++i)
#pragma unroll
    for (int j = 0; j < 4; ++j)
      acc[i][j] = (v8f){0.f, 0.f, 0.f, 0.f, 0.f, 0.f, 0.f, 0.f};

#if !HAVE_ASYNC
  uint4 breg[4];
#endif
#if !HAVE_TDM
  uint4 areg[4];
#endif

  // ---- prologue: stage chunk 0 into buffer 0 ----
#if HAVE_TDM
  if (wave == 0) tdm_issue_a(0, 0);
#else
  stage_a_load(0, areg);
  stage_a_store(0, areg);
#endif
#if HAVE_ASYNC
  stage_b_async(0, 0);
#else
  stage_b_load(0, breg);
  stage_b_store(0, breg);
#endif

  int buf = 0;
  for (int c = 0; c < NCHUNK; ++c) {
#if HAVE_TDM
    if (wave == 0) __builtin_amdgcn_s_wait_tensorcnt(0);
#endif
#if HAVE_ASYNC
    wait_asynccnt0();
#endif
    __syncthreads();   // buf published to all waves; buf^1 free everywhere

    if (c + 1 < NCHUNK) {
#if HAVE_TDM
      if (wave == 0) tdm_issue_a(c + 1, buf ^ 1);
#else
      stage_a_load(c + 1, areg);
#endif
#if HAVE_ASYNC
      stage_b_async(c + 1, buf ^ 1);   // DMA into idle buffer under the WMMAs
#else
      stage_b_load(c + 1, breg);
#endif
    }

    // ---- fragments for chunk c ----
    v16bf aHi[2], aLo[2], bHi[4], bLo[4];
#pragma unroll
    for (int rt = 0; rt < 2; ++rt) {
      const int rb = waveRow * 32 + rt * 16;
      aHi[rt] = load_frag_a(&lsA[buf][0],     rb, lane);
      aLo[rt] = load_frag_a(&lsA[buf][ASLOT], rb, lane);
    }
#pragma unroll
    for (int ct = 0; ct < 4; ++ct) {
      const int cb = waveCol * 64 + ct * 16;
      bHi[ct] = load_frag_b(&lsB[buf][0],     cb, lane);
      bLo[ct] = load_frag_b(&lsB[buf][BSLOT], cb, lane);
    }

    // ---- bf16x3: hi*hi + hi*lo + lo*hi, f32 accumulate ----
#pragma unroll
    for (int rt = 0; rt < 2; ++rt) {
#pragma unroll
      for (int ct = 0; ct < 4; ++ct) {
        acc[rt][ct] = __builtin_amdgcn_wmma_f32_16x16x32_bf16(
            false, aHi[rt], false, bHi[ct], (short)0, acc[rt][ct], false, false);
        acc[rt][ct] = __builtin_amdgcn_wmma_f32_16x16x32_bf16(
            false, aHi[rt], false, bLo[ct], (short)0, acc[rt][ct], false, false);
        acc[rt][ct] = __builtin_amdgcn_wmma_f32_16x16x32_bf16(
            false, aLo[rt], false, bHi[ct], (short)0, acc[rt][ct], false, false);
      }
    }

    if (c + 1 < NCHUNK) {
#if !HAVE_TDM
      stage_a_store(buf ^ 1, areg);
#endif
#if !HAVE_ASYNC
      stage_b_store(buf ^ 1, breg);
#endif
    }
    buf ^= 1;
  }

  // ---- epilogue: scatter C tiles into NCHW output ----
  const int n16 = lane & 15;
  const int mh  = lane >> 4;
#pragma unroll
  for (int rt = 0; rt < 2; ++rt) {
    const int mBase = blockRow * 128 + waveRow * 32 + rt * 16 + mh * 8;
#pragma unroll
    for (int ct = 0; ct < 4; ++ct) {
      const int pG  = blockCol * 128 + waveCol * 64 + ct * 16 + n16;
      const int nI  = pG / HW;
      const int phw = pG - nI * HW;
      float* op = out + (size_t)nI * COUT * HW + phw;
#pragma unroll
      for (int v = 0; v < 8; ++v)
        op[(size_t)(mBase + v) * HW] = acc[rt][ct][v];
    }
  }
}

extern "C" void kernel_launch(void* const* d_in, const int* in_sizes, int n_in,
                              void* d_out, int out_size, void* d_ws, size_t ws_size,
                              hipStream_t stream) {
  const float* x = (const float*)d_in[0];   // [32,256,56,56] f32
  const float* w = (const float*)d_in[1];   // [256,256,3,3]  f32
  float* out = (float*)d_out;               // [32,256,56,56] f32
  (void)in_sizes; (void)n_in; (void)out_size; (void)ws_size;

  // workspace carve (needs ~105.2 MB):
  //   xs  : [2][NPIX][CIN] bf16  = 102.76 MB
  //   wsp : [2][COUT][KTOT] bf16 =   2.36 MB
  //   zpad: 64 B zero block for async OOB redirection
  unsigned short* xs   = (unsigned short*)d_ws;
  unsigned short* wsp  = xs + (size_t)2 * XSP;
  unsigned short* zpad = wsp + (size_t)2 * WSP;

  prep_x<<<dim3(NPIX), dim3(256), 0, stream>>>(x, xs);
  prep_w<<<dim3(COUT), dim3(256), 0, stream>>>(w, wsp, zpad);
  conv_gemm<<<dim3(784, 2), dim3(256), 0, stream>>>(xs, wsp, zpad, out);
}
